// GATEncoder_55748675502367
// MI455X (gfx1250) — matrix operand order repaired and checked by reading, not compiled, not run on hardware
//
#include <hip/hip_runtime.h>

typedef __attribute__((ext_vector_type(16))) _Float16 v16h;
typedef __attribute__((ext_vector_type(8)))  float    v8f;

#define NEG_SLOPE 0.2f
#define SM_EPS    1e-16f

// ---------------------------------------------------------------------------
// Native device-scope f32 atomic add (GLOBAL_ATOMIC_ADD_F32, no return).
// ---------------------------------------------------------------------------
__device__ __forceinline__ void atomic_add_f32(float* addr, float val) {
  asm volatile("global_atomic_add_f32 %0, %1, off scope:SCOPE_DEV"
               :: "v"(addr), "v"(val) : "memory");
}

// Order-preserving float -> uint key (for atomicMax-based segment max).
__device__ __forceinline__ unsigned f32_key(float f) {
  unsigned b = __float_as_uint(f);
  return (b & 0x80000000u) ? ~b : (b | 0x80000000u);
}
__device__ __forceinline__ float key_f32(unsigned k) {
  unsigned b = (k & 0x80000000u) ? (k ^ 0x80000000u) : ~k;
  return __uint_as_float(b);
}

// ---------------------------------------------------------------------------
// f32 -> f16 conversion
// ---------------------------------------------------------------------------
__global__ void k_f32_to_f16(const float* __restrict__ in,
                             _Float16* __restrict__ out, int n) {
  int i = blockIdx.x * blockDim.x + threadIdx.x;
  if (i < n) out[i] = (_Float16)in[i];
}

// ---------------------------------------------------------------------------
// H[M,64] = A[M,K](f16) @ W[K,64](f32), WMMA f32_16x16x32_f16.
// 256 threads = 8 waves; each wave computes one 16x64 strip.
// W is staged into LDS pre-swizzled into per-fragment, per-lane contiguous
// 32-byte runs so each B fragment is two ds_load_b128.
//
// Fragment (ks, nt), lane l, element i maps to:
//   k = ks*32 + (i<8 ? (l>>4)*8 + i : 16 + (l>>4)*8 + (i-8)),  n = nt*16 + (l&15)
// stored at ldsWf[ ((ks*4+nt)*32 + l)*16 + i ].
// ---------------------------------------------------------------------------
template<int K>
__global__ __launch_bounds__(256) void k_gemm_wmma(
    const _Float16* __restrict__ A, const float* __restrict__ W,
    float* __restrict__ H, int M) {
  __shared__ _Float16 ldsWf[K * 64];

  // Cooperative staging with fragment swizzle (+ f32 -> f16 convert).
  for (int idx = threadIdx.x; idx < K * 64; idx += 256) {
    const int frag = idx >> 9;          // 512 halfs per fragment
    const int rem  = idx & 511;
    const int l    = rem >> 4;          // lane
    const int i    = rem & 15;          // element
    const int ks   = frag >> 2;
    const int nt   = frag & 3;
    const int k    = ks * 32 + (i < 8 ? (l >> 4) * 8 + i : 16 + (l >> 4) * 8 + (i - 8));
    const int n    = nt * 16 + (l & 15);
    ldsWf[idx] = (_Float16)W[k * 64 + n];
  }
  __syncthreads();

  const int lane = threadIdx.x & 31;
  const int wave = threadIdx.x >> 5;
  const int tile = blockIdx.x * 8 + wave;
  if (tile >= (M >> 4)) return;

  const int hi  = lane >> 4;    // half-wave select
  const int l15 = lane & 15;
  const int row = (tile << 4) + l15;

  const _Float16* fragBase = ldsWf + lane * 16;   // + frag*512

  v8f c0 = {}, c1 = {}, c2 = {}, c3 = {};

#pragma unroll
  for (int ks = 0; ks < K / 32; ++ks) {
    // A fragment: lanes 0-15 hold K k0..k0+7 and k0+16..k0+23 (k0=ks*32),
    // lanes 16-31 the same shifted by 8.
    v16h a;
    const int k0 = ks * 32 + hi * 8;
    *(uint4*)&a         = *(const uint4*)(A + (size_t)row * K + k0);
    *(((uint4*)&a) + 1) = *(const uint4*)(A + (size_t)row * K + k0 + 16);

    // B fragments: contiguous 32 bytes per lane in swizzled LDS image.
    const _Float16* fb = fragBase + ks * 4 * 512;
    v16h b0, b1, b2, b3;
    *(uint4*)&b0         = *(const uint4*)(fb + 0 * 512);
    *(((uint4*)&b0) + 1) = *(const uint4*)(fb + 0 * 512 + 8);
    *(uint4*)&b1         = *(const uint4*)(fb + 1 * 512);
    *(((uint4*)&b1) + 1) = *(const uint4*)(fb + 1 * 512 + 8);
    *(uint4*)&b2         = *(const uint4*)(fb + 2 * 512);
    *(((uint4*)&b2) + 1) = *(const uint4*)(fb + 2 * 512 + 8);
    *(uint4*)&b3         = *(const uint4*)(fb + 3 * 512);
    *(((uint4*)&b3) + 1) = *(const uint4*)(fb + 3 * 512 + 8);

    c0 = __builtin_amdgcn_wmma_f32_16x16x32_f16(false, a, false, b0, (short)0, c0, false, false);
    c1 = __builtin_amdgcn_wmma_f32_16x16x32_f16(false, a, false, b1, (short)0, c1, false, false);
    c2 = __builtin_amdgcn_wmma_f32_16x16x32_f16(false, a, false, b2, (short)0, c2, false, false);
    c3 = __builtin_amdgcn_wmma_f32_16x16x32_f16(false, a, false, b3, (short)0, c3, false, false);
  }

  // C/D layout: VGPR j -> M = j (lanes 0-15) / 8+j (lanes 16-31), N = lane&15.
  float* out = H + (size_t)(tile << 4) * 64;
#pragma unroll
  for (int j = 0; j < 8; ++j) {
    const int r = j + hi * 8;
    out[r * 64 +      l15] = c0[j];
    out[r * 64 + 16 + l15] = c1[j];
    out[r * 64 + 32 + l15] = c2[j];
    out[r * 64 + 48 + l15] = c3[j];
  }
}

// ---------------------------------------------------------------------------
// Per-node attention coefficients: a_src[n] = h[n].att_src, a_dst likewise.
// ---------------------------------------------------------------------------
__global__ void k_attn_coeff(const float* __restrict__ H,
                             const float* __restrict__ att_s,
                             const float* __restrict__ att_d,
                             float* __restrict__ as_, float* __restrict__ ad_,
                             int N) {
  int n = blockIdx.x * blockDim.x + threadIdx.x;
  if (n >= N) return;
  const float4* h4 = (const float4*)(H + (size_t)n * 64);
  const float4* s4 = (const float4*)att_s;
  const float4* d4 = (const float4*)att_d;
  float s = 0.f, d = 0.f;
#pragma unroll
  for (int i = 0; i < 16; ++i) {
    float4 h = h4[i], sv = s4[i], dv = d4[i];
    s += h.x * sv.x + h.y * sv.y + h.z * sv.z + h.w * sv.w;
    d += h.x * dv.x + h.y * dv.y + h.z * dv.z + h.w * dv.w;
  }
  as_[n] = s; ad_[n] = d;
}

__device__ __forceinline__ void edge_pair(const long long* ei, int e, int E,
                                          int& src, int& dst) {
  if (e < E) { src = (int)ei[e]; dst = (int)ei[(size_t)E + e]; }
  else       { src = dst = e - E; }
}

// ---------------------------------------------------------------------------
// Pass 1: segment max of leaky-relu logits via atomicMax on ordered keys.
// ---------------------------------------------------------------------------
__global__ void k_edge_max(const long long* __restrict__ ei,
                           const float* __restrict__ as_,
                           const float* __restrict__ ad_,
                           unsigned* __restrict__ maxkey, int E, int N) {
  int e = blockIdx.x * blockDim.x + threadIdx.x;
  if (e >= E + N) return;
  int src, dst; edge_pair(ei, e, E, src, dst);
  float v = as_[src] + ad_[dst];
  v = v > 0.f ? v : NEG_SLOPE * v;
  atomicMax(maxkey + dst, f32_key(v));
}

// ---------------------------------------------------------------------------
// Pass 2: p = exp(v - m); store p, atomic-add segment sum.
// ---------------------------------------------------------------------------
__global__ void k_edge_expsum(const long long* __restrict__ ei,
                              const float* __restrict__ as_,
                              const float* __restrict__ ad_,
                              const unsigned* __restrict__ maxkey,
                              float* __restrict__ pbuf,
                              float* __restrict__ ssum, int E, int N) {
  int e = blockIdx.x * blockDim.x + threadIdx.x;
  if (e >= E + N) return;
  int src, dst; edge_pair(ei, e, E, src, dst);
  float v = as_[src] + ad_[dst];
  v = v > 0.f ? v : NEG_SLOPE * v;
  float m = key_f32(maxkey[dst]);
  if ((__float_as_uint(m) & 0x7F800000u) == 0x7F800000u) m = 0.f;  // !finite -> 0
  float p = __expf(v - m);
  pbuf[e] = p;
  atomic_add_f32(ssum + dst, p);
}

// ---------------------------------------------------------------------------
// Pass 3: agg[dst] += alpha * h[src]. One wave per edge, 2 features per lane.
// ---------------------------------------------------------------------------
__global__ __launch_bounds__(256) void k_edge_agg(
    const long long* __restrict__ ei, const float* __restrict__ pbuf,
    const float* __restrict__ ssum, const float* __restrict__ H,
    float* __restrict__ agg, int E, int N) {
  int lane = threadIdx.x & 31;
  int e = (blockIdx.x * 256 + threadIdx.x) >> 5;
  if (e >= E + N) return;
  int src, dst; edge_pair(ei, e, E, src, dst);
  float alpha = pbuf[e] / (ssum[dst] + SM_EPS);
  float2 h = *(const float2*)(H + (size_t)src * 64 + lane * 2);
  float* o = agg + (size_t)dst * 64 + lane * 2;
  atomic_add_f32(o,     alpha * h.x);
  atomic_add_f32(o + 1, alpha * h.y);
}

// ---------------------------------------------------------------------------
// Epilogues
// ---------------------------------------------------------------------------
__global__ void k_bias_relu_f16(const float* __restrict__ agg,
                                const float* __restrict__ b,
                                _Float16* __restrict__ xh, int n) {
  int i = blockIdx.x * blockDim.x + threadIdx.x;
  if (i >= n) return;
  float v = agg[i] + b[i & 63];
  xh[i] = (_Float16)(v > 0.f ? v : 0.f);
}

__global__ void k_bias_out(const float* __restrict__ agg,
                           const float* __restrict__ b,
                           float* __restrict__ out, int n) {
  int i = blockIdx.x * blockDim.x + threadIdx.x;
  if (i >= n) return;
  out[i] = agg[i] + b[i & 63];
}

// ---------------------------------------------------------------------------
// Host orchestration
// ---------------------------------------------------------------------------
extern "C" void kernel_launch(void* const* d_in, const int* in_sizes, int n_in,
                              void* d_out, int out_size, void* d_ws, size_t ws_size,
                              hipStream_t stream) {
  const float*     x    = (const float*)d_in[0];
  const long long* ei   = (const long long*)d_in[1];
  const float*     W1   = (const float*)d_in[2];
  const float*     as1  = (const float*)d_in[3];
  const float*     ad1  = (const float*)d_in[4];
  const float*     b1   = (const float*)d_in[5];
  const float*     W2   = (const float*)d_in[6];
  const float*     as2  = (const float*)d_in[7];
  const float*     ad2  = (const float*)d_in[8];
  const float*     b2   = (const float*)d_in[9];
  float* out = (float*)d_out;

  const int N  = in_sizes[0] / 128;   // 50000
  const int E  = in_sizes[1] / 2;     // 800000
  const int EN = E + N;

  // workspace carve-up (all 16B aligned)
  char* w = (char*)d_ws;
  _Float16* xh  = (_Float16*)w; w += (size_t)N * 128 * 2;
  _Float16* x1h = (_Float16*)w; w += (size_t)N * 64 * 2;
  float*    H   = (float*)w;    w += (size_t)N * 64 * 4;
  float*    agg = (float*)w;    w += (size_t)N * 64 * 4;
  float*    cs  = (float*)w;    w += (size_t)N * 4;    // a_src
  float*    cd  = (float*)w;    w += (size_t)N * 4;    // a_dst
  unsigned* mk  = (unsigned*)w; w += (size_t)N * 4;    // max keys
  float*    ss  = (float*)w;    w += (size_t)N * 4;    // softmax denom
  float*    pb  = (float*)w;    w += (size_t)EN * 4;   // exp(e - m)

  const int T = 256;
  dim3 blk(T);
  const int gTiles = ((N >> 4) + 7) / 8;
  const int gNode  = (N + T - 1) / T;
  const int gEdge  = (EN + T - 1) / T;
  const int gAgg   = (EN * 32 + T - 1) / T;
  const int gFeat  = (N * 64 + T - 1) / T;

  // ---- layer 1 ----
  k_f32_to_f16<<<(N * 128 + T - 1) / T, blk, 0, stream>>>(x, xh, N * 128);
  hipMemsetAsync(agg, 0, (size_t)N * 64 * 4, stream);
  hipMemsetAsync(mk,  0, (size_t)N * 4, stream);   // key 0 == -inf sentinel
  hipMemsetAsync(ss,  0, (size_t)N * 4, stream);

  k_gemm_wmma<128><<<gTiles, blk, 0, stream>>>(xh, W1, H, N);
  k_attn_coeff<<<gNode, blk, 0, stream>>>(H, as1, ad1, cs, cd, N);
  k_edge_max<<<gEdge, blk, 0, stream>>>(ei, cs, cd, mk, E, N);
  k_edge_expsum<<<gEdge, blk, 0, stream>>>(ei, cs, cd, mk, pb, ss, E, N);
  k_edge_agg<<<gAgg, blk, 0, stream>>>(ei, pb, ss, H, agg, E, N);
  k_bias_relu_f16<<<gFeat, blk, 0, stream>>>(agg, b1, x1h, N * 64);

  // ---- layer 2 ----
  hipMemsetAsync(agg, 0, (size_t)N * 64 * 4, stream);
  hipMemsetAsync(mk,  0, (size_t)N * 4, stream);
  hipMemsetAsync(ss,  0, (size_t)N * 4, stream);

  k_gemm_wmma<64><<<gTiles, blk, 0, stream>>>(x1h, W2, H, N);
  k_attn_coeff<<<gNode, blk, 0, stream>>>(H, as2, ad2, cs, cd, N);
  k_edge_max<<<gEdge, blk, 0, stream>>>(ei, cs, cd, mk, E, N);
  k_edge_expsum<<<gEdge, blk, 0, stream>>>(ei, cs, cd, mk, pb, ss, E, N);
  k_edge_agg<<<gAgg, blk, 0, stream>>>(ei, pb, ss, H, agg, E, N);
  k_bias_out<<<gFeat, blk, 0, stream>>>(agg, b2, out, N * 64);
}
